// FeatureWiseAttention_21895743275408
// MI455X (gfx1250) — matrix operand order repaired and checked by reading, not compiled, not run on hardware
//
#include <hip/hip_runtime.h>

typedef __attribute__((ext_vector_type(16))) __bf16 v16bf;
typedef __attribute__((ext_vector_type(8)))  float  v8f;
typedef __attribute__((ext_vector_type(8)))  unsigned v8u;

#define F_DIM   1024
#define M_TILE  32
#define THREADS 256
#define N_WAVES 8
#define N_TILES 8      // 8 * 16 = 128 columns per wave
#define K_STEP  32
#define K_ITERS (F_DIM / K_STEP)

union BF16x16 {
  v16bf v;
  unsigned u[8];
  v8u w;
};

// word = { bf16(hi)[31:16], bf16(lo)[15:0] } by byte-select (truncating f32->bf16)
static __device__ __forceinline__ unsigned pack_bf16(float lo, float hi) {
  return __builtin_amdgcn_perm(__builtin_bit_cast(unsigned, hi),
                               __builtin_bit_cast(unsigned, lo),
                               0x07060302u);
}

// Convert 8 contiguous floats (one 32B load) into 4 packed bf16 words.
static __device__ __forceinline__ void pack8(BF16x16& d, int base, v8f f) {
  d.u[base + 0] = pack_bf16(f[0], f[1]);
  d.u[base + 1] = pack_bf16(f[2], f[3]);
  d.u[base + 2] = pack_bf16(f[4], f[5]);
  d.u[base + 3] = pack_bf16(f[6], f[7]);
}

// f32 -> bf16 with round-to-nearest-even (used once for W, so accuracy > speed)
static __device__ __forceinline__ unsigned rne_bf16_word(float lo, float hi) {
  unsigned ul = __builtin_bit_cast(unsigned, lo);
  unsigned uh = __builtin_bit_cast(unsigned, hi);
  ul += 0x7FFFu + ((ul >> 16) & 1u);
  uh += 0x7FFFu + ((uh >> 16) & 1u);
  return (ul >> 16) | (uh & 0xFFFF0000u);
}

// One-shot W conversion: [1024,1024] fp32 -> bf16 words in workspace.
__global__ __launch_bounds__(256)
void conv_w_kernel(const float* __restrict__ W, unsigned* __restrict__ Wb) {
  const int i = blockIdx.x * 256 + threadIdx.x;   // packed-word index (2 floats)
  const float2 f = ((const float2*)W)[i];
  Wb[i] = rne_bf16_word(f.x, f.y);
}

template <bool WBF16>
__global__ __launch_bounds__(THREADS)
void fwa_kernel(const float* __restrict__ x, const float* __restrict__ Wf,
                const unsigned* __restrict__ Wb, const float* __restrict__ bias,
                float* __restrict__ out) {
  __shared__ float s_part[M_TILE][N_WAVES];
  __shared__ float s_row[M_TILE];

  const int tid  = threadIdx.x;
  const int wave = tid >> 5;       // wave32
  const int lane = tid & 31;
  const int hi   = lane >> 4;      // which 16-lane half
  const int ln   = lane & 15;
  const int rowBase = blockIdx.x * M_TILE;
  const int nBase   = wave * (N_TILES * 16);

  v8f acc0[N_TILES], acc1[N_TILES];
  const v8f vzero = {0.f, 0.f, 0.f, 0.f, 0.f, 0.f, 0.f, 0.f};
#pragma unroll
  for (int t = 0; t < N_TILES; ++t) { acc0[t] = vzero; acc1[t] = vzero; }

  // A fragment rows: M = ln for both lane halves (lanes 0-15: K 0..15,
  // lanes 16-31: K 16..31, contiguous, matching B).
  const float* xa0 = x + (size_t)(rowBase + ln) * F_DIM + hi * 16;
  const float* xa1 = xa0 + (size_t)16 * F_DIM;   // folds to +65536B immediate
  // B stream: one base pointer; every tile of a k-step is a compile-time
  // immediate offset away (t*16 rows * 2048B = t*32768B, fits i24 offset).
  const unsigned* wk =
      Wb + (((size_t)(nBase + ln) * F_DIM + hi * 16) >> 1);
  const float* wkf =
      Wf + (size_t)(nBase + ln) * F_DIM + hi * 16;

#pragma unroll 4
  for (int it = 0; it < K_ITERS; ++it) {
    BF16x16 a0, a1;
    pack8(a0, 0, *(const v8f*)(xa0));
    pack8(a0, 4, *(const v8f*)(xa0 + 8));
    pack8(a1, 0, *(const v8f*)(xa1));
    pack8(a1, 4, *(const v8f*)(xa1 + 8));

#pragma unroll
    for (int t = 0; t < N_TILES; ++t) {
      // B fragment: lane = N (ln), lanes 0-15 hold K=k0..k0+15, 16-31 hold K+16.
      BF16x16 bt;
      if constexpr (WBF16) {
        // bf16 W: 16 contiguous bf16 = one 32B-aligned load; memory pair order
        // == VGPR word order; per-tile delta is an immediate offset.
        bt.w = *(const v8u*)(wk + t * (16 * F_DIM / 2));
      } else {
        const float* wr = wkf + (size_t)t * 16 * F_DIM;
        pack8(bt, 0, *(const v8f*)(wr));
        pack8(bt, 4, *(const v8f*)(wr + 8));
      }
      acc0[t] = __builtin_amdgcn_wmma_f32_16x16x32_bf16(
          false, a0.v, false, bt.v, (short)0, acc0[t], false, false);
      acc1[t] = __builtin_amdgcn_wmma_f32_16x16x32_bf16(
          false, a1.v, false, bt.v, (short)0, acc1[t], false, false);
    }
    xa0 += K_STEP;
    xa1 += K_STEP;
    wk  += K_STEP / 2;   // 32 bf16 = 16 words = 64B per k-step
    wkf += K_STEP;
  }

  // Add bias (per output feature / column).
#pragma unroll
  for (int t = 0; t < N_TILES; ++t) {
    const float bv = bias[nBase + t * 16 + ln];
#pragma unroll
    for (int r = 0; r < 8; ++r) { acc0[t][r] += bv; acc1[t][r] += bv; }
  }

  // C layout: element r of acc belongs to row (s*16 + hi*8 + r), col = nBase+t*16+ln.
  // ---- per-row max: reduce over t, then over the 16-lane half, then across waves.
#pragma unroll
  for (int r = 0; r < 8; ++r) {
    float m0 = acc0[0][r], m1 = acc1[0][r];
#pragma unroll
    for (int t = 1; t < N_TILES; ++t) {
      m0 = fmaxf(m0, acc0[t][r]);
      m1 = fmaxf(m1, acc1[t][r]);
    }
#pragma unroll
    for (int off = 1; off < 16; off <<= 1) {
      m0 = fmaxf(m0, __shfl_xor(m0, off, 32));
      m1 = fmaxf(m1, __shfl_xor(m1, off, 32));
    }
    if (ln == 0) {
      s_part[hi * 8 + r][wave]      = m0;
      s_part[16 + hi * 8 + r][wave] = m1;
    }
  }
  __syncthreads();
  if (tid < M_TILE) {
    float m = s_part[tid][0];
#pragma unroll
    for (int w = 1; w < N_WAVES; ++w) m = fmaxf(m, s_part[tid][w]);
    s_row[tid] = m;
  }
  __syncthreads();

  // ---- exp in place + per-row sums.
#pragma unroll
  for (int r = 0; r < 8; ++r) {
    const float mx0 = s_row[hi * 8 + r];
    const float mx1 = s_row[16 + hi * 8 + r];
    float s0 = 0.f, s1 = 0.f;
#pragma unroll
    for (int t = 0; t < N_TILES; ++t) {
      const float e0 = __expf(acc0[t][r] - mx0);
      const float e1 = __expf(acc1[t][r] - mx1);
      acc0[t][r] = e0; acc1[t][r] = e1;
      s0 += e0; s1 += e1;
    }
#pragma unroll
    for (int off = 1; off < 16; off <<= 1) {
      s0 += __shfl_xor(s0, off, 32);
      s1 += __shfl_xor(s1, off, 32);
    }
    if (ln == 0) {
      s_part[hi * 8 + r][wave]      = s0;
      s_part[16 + hi * 8 + r][wave] = s1;
    }
  }
  __syncthreads();
  if (tid < M_TILE) {
    float s = s_part[tid][0];
#pragma unroll
    for (int w = 1; w < N_WAVES; ++w) s += s_part[tid][w];
    s_row[tid] = 1.0f / s;
  }
  __syncthreads();

  // ---- out = x * (1 + attn); x re-read (hot in L2/WGP$).
#pragma unroll
  for (int r = 0; r < 8; ++r) {
    const int row0 = rowBase + hi * 8 + r;
    const int row1 = row0 + 16;
    const float inv0 = s_row[hi * 8 + r];
    const float inv1 = s_row[16 + hi * 8 + r];
#pragma unroll
    for (int t = 0; t < N_TILES; ++t) {
      const int col = nBase + t * 16 + ln;
      const float xv0 = x[(size_t)row0 * F_DIM + col];
      const float xv1 = x[(size_t)row1 * F_DIM + col];
      out[(size_t)row0 * F_DIM + col] = xv0 * (1.0f + acc0[t][r] * inv0);
      out[(size_t)row1 * F_DIM + col] = xv1 * (1.0f + acc1[t][r] * inv1);
    }
  }
}

extern "C" void kernel_launch(void* const* d_in, const int* in_sizes, int n_in,
                              void* d_out, int out_size, void* d_ws, size_t ws_size,
                              hipStream_t stream) {
  (void)n_in; (void)out_size;
  const float* x    = (const float*)d_in[0];
  const float* W    = (const float*)d_in[1];
  const float* bias = (const float*)d_in[2];
  float* out = (float*)d_out;

  const int M = in_sizes[0] / F_DIM;        // 16*2048 = 32768 rows
  const int blocks = M / M_TILE;            // 1024 blocks
  const size_t wbfBytes = (size_t)F_DIM * F_DIM * 2;

  if (d_ws != nullptr && ws_size >= wbfBytes) {
    // One-shot fp32->bf16 conversion of W into workspace (RNE), then the
    // bf16-B GEMM path: half the L2 bytes and no per-iteration B conversion.
    const int words = F_DIM * F_DIM / 2;            // 524288
    conv_w_kernel<<<dim3(words / 256), dim3(256), 0, stream>>>(
        W, (unsigned*)d_ws);
    fwa_kernel<true><<<dim3(blocks), dim3(THREADS), 0, stream>>>(
        x, W, (const unsigned*)d_ws, bias, out);
  } else {
    fwa_kernel<false><<<dim3(blocks), dim3(THREADS), 0, stream>>>(
        x, W, nullptr, bias, out);
  }
}
